// DualTowerHypergraphSummariser_34952443855221
// MI455X (gfx1250) — matrix operand order repaired and checked by reading, not compiled, not run on hardware
//
#include <hip/hip_runtime.h>

// ---------------------------------------------------------------------------
// DualTowerHypergraphSummariser for MI455X (gfx1250, wave32, WMMA)
// Big per-step GEMMs: bf16 WMMA, f32 accumulate, async global->LDS double
// buffering (ASYNCcnt, one barrier per K-chunk), K-chunk 64, dual-problem
// launches via blockIdx.z. Small/elementwise work in exact f32 (split-K).
// ---------------------------------------------------------------------------

typedef __bf16 bf16_t;
typedef __attribute__((ext_vector_type(16))) __bf16 v16bf;
typedef __attribute__((ext_vector_type(8)))  float  v8f;

#define B_  8
#define S_  48
#define N_  100
#define D_  1024
#define BN_ 800      // B_*N_
#define BS_ 384      // B_*S_

// ---- CDNA5 async global->LDS (tracked by ASYNCcnt), ISA 08_async_tensor ----
__device__ __forceinline__ void async_load_b128(unsigned lds_byte_addr,
                                                const void* gaddr) {
  asm volatile("global_load_async_to_lds_b128 %0, %1, off"
               :: "v"(lds_byte_addr),
                  "v"((unsigned long long)(__SIZE_TYPE__)gaddr)
               : "memory");
}
__device__ __forceinline__ void wait_asynccnt_0() {
#if __has_builtin(__builtin_amdgcn_s_wait_asynccnt)
  __builtin_amdgcn_s_wait_asynccnt(0);
#else
  asm volatile("s_wait_asynccnt 0x0" ::: "memory");
#endif
}

// ---------------------------------------------------------------------------
// WMMA GEMM: C[M,N] = A[M,K](bf16,row-major) * Bt[N,K](bf16,row-major)^T (+bias)
// Block = 128 threads (4 waves). Tile 128(M) x 64(N), K-chunk 64 (2 WMMA
// sub-chunks), double-buffered async loads, ONE barrier per chunk:
//   wait_asynccnt(0) -> barrier -> issue(next chunk, other buf) -> compute.
// The barrier proves (a) all waves' chunk-i data landed, (b) all waves are
// done computing chunk i-1, so overwriting buf[(i+1)&1] is safe.
// Wave w owns rows [w*32,w*32+32): 16 v_wmma per chunk; all fragments are
// preloaded into distinct registers so DS waits are partial, WMMAs b2b.
// blockIdx.z picks problem {0,1}.
// ---------------------------------------------------------------------------
__global__ __launch_bounds__(128) void wmma_gemm_kernel(
    const bf16_t* __restrict__ A0, const bf16_t* __restrict__ A1,
    const bf16_t* __restrict__ B0, const bf16_t* __restrict__ B1,
    const float* __restrict__ bias0, const float* __restrict__ bias1,
    float* __restrict__ C0, float* __restrict__ C1,
    int M, int N, int K) {
  __shared__ bf16_t As[2][128][72];   // stride 72 elems = 144B: 16B aligned,
  __shared__ bf16_t Bs[2][64][72];    // bank pattern 36*r mod 64 conflict-free
  const int z = blockIdx.z;
  const bf16_t* __restrict__ A   = z ? A1 : A0;
  const bf16_t* __restrict__ Bt  = z ? B1 : B0;
  const float* __restrict__ bias = z ? bias1 : bias0;
  float* __restrict__ C          = z ? C1 : C0;

  const int tid  = threadIdx.x;
  const int wave = tid >> 5;
  const int lane = tid & 31;
  const int hlf  = lane >> 4;         // lane half (K interleave select)
  const int l15  = lane & 15;
  const int m0 = blockIdx.y * 128;
  const int n0 = blockIdx.x * 64;

  union { v8f v; float f[8]; } acc[2][4];
#pragma unroll
  for (int a = 0; a < 2; ++a)
#pragma unroll
    for (int jt = 0; jt < 4; ++jt)
#pragma unroll
      for (int i = 0; i < 8; ++i) acc[a][jt].f[i] = 0.0f;

  const int trow = tid >> 3;          // 0..15
  const int tq   = tid & 7;           // 16B quad within a 64-elem K-chunk row

  const unsigned aoff = (unsigned)(__SIZE_TYPE__)(void*)&As[0][0][0];
  const unsigned boff = (unsigned)(__SIZE_TYPE__)(void*)&Bs[0][0][0];

  // Issue one K-chunk of async global->LDS traffic: 12 ops/wave
  // (A: 128 rows x 64 k -> 8 ops, B: 64 rows x 64 k -> 4 ops).
  auto issue = [&](int k0, int bufid) {
    const unsigned ab = aoff + (unsigned)bufid * (128u * 72u * 2u);
    const unsigned bb = boff + (unsigned)bufid * (64u * 72u * 2u);
#pragma unroll
    for (int i = 0; i < 8; ++i) {
      const int row = trow + i * 16;
      int gm = m0 + row;                   // clamp instead of branch: dup row
      gm = gm < M ? gm : (M - 1);          // M-1; OOB C rows are never stored
      async_load_b128(ab + (unsigned)(row * 144 + tq * 16),
                      A + (size_t)gm * K + k0 + tq * 8);
    }
#pragma unroll
    for (int i = 0; i < 4; ++i) {
      const int row = trow + i * 16;
      async_load_b128(bb + (unsigned)(row * 144 + tq * 16),
                      Bt + (size_t)(n0 + row) * K + k0 + tq * 8);
    }
  };

  const int nk = K >> 6;
  issue(0, 0);
  for (int ki = 0; ki < nk; ++ki) {
    const int cur = ki & 1;
    wait_asynccnt_0();                 // own chunk-ki DMAs landed in LDS
    __syncthreads();                   // all waves' data landed; all waves
                                       // finished computing chunk ki-1
    if (ki + 1 < nk) issue((ki + 1) << 6, cur ^ 1);  // prefetch during compute

#pragma unroll
    for (int c = 0; c < 2; ++c) {      // two 32-wide WMMA sub-chunks
      const int kbase = c * 32;
      // Preload ALL fragments into distinct registers -> partial DS waits.
      // A fragments: 16x32 bf16, ISA layout (lane=row&15; K pairs by half)
      union { v16bf v; unsigned u[8]; } afrag[2];
#pragma unroll
      for (int a = 0; a < 2; ++a) {
        const int ar = wave * 32 + a * 16 + l15;
#pragma unroll
        for (int j = 0; j < 8; ++j) {
          const int kb = kbase + ((j >> 2) << 4) + hlf * 8 + ((j & 3) << 1);
          afrag[a].u[j] = *(const unsigned*)(&As[cur][ar][kb]);
        }
      }
      // B fragments: 32x16 bf16 (lane=col&15; K pairs by half, same as A)
      union { v16bf v; unsigned u[8]; } bfrag[4];
#pragma unroll
      for (int jt = 0; jt < 4; ++jt) {
        const int bc = jt * 16 + l15;
#pragma unroll
        for (int j = 0; j < 8; ++j) {
          const int kk = kbase + ((j >> 2) << 4) + hlf * 8 + ((j & 3) << 1);
          bfrag[jt].u[j] = *(const unsigned*)(&Bs[cur][bc][kk]);
        }
      }
#pragma unroll
      for (int jt = 0; jt < 4; ++jt)
#pragma unroll
        for (int a = 0; a < 2; ++a)
          acc[a][jt].v = __builtin_amdgcn_wmma_f32_16x16x32_bf16(
              false, afrag[a].v, false, bfrag[jt].v, (short)0, acc[a][jt].v,
              false, false);
    }
  }

#pragma unroll
  for (int a = 0; a < 2; ++a)
#pragma unroll
    for (int jt = 0; jt < 4; ++jt) {
      const int c = n0 + jt * 16 + l15;
      const float bv = bias ? bias[c] : 0.0f;
#pragma unroll
      for (int v = 0; v < 8; ++v) {
        const int m = m0 + wave * 32 + a * 16 + hlf * 8 + v;  // M = v + 8*half
        if (m < M) C[(size_t)m * N + c] = acc[a][jt].f[v] + bv;
      }
    }
}

// ---------------------------------------------------------------------------
// Setup kernels
// ---------------------------------------------------------------------------
__global__ void stream_softmax_kernel(const float* __restrict__ w, float* __restrict__ sm) {
  const float a = w[0], b = w[1], c = w[2];
  const float m = fmaxf(a, fmaxf(b, c));
  const float ea = expf(a - m), eb = expf(b - m), ec = expf(c - m);
  const float t = ea + eb + ec;
  sm[0] = ea / t; sm[1] = eb / t; sm[2] = ec / t;
}

// Wt[n*K+k] = bf16(W[k*N+n])   (W is [K,N] row-major; Wt is [N,K] row-major)
__global__ void transpose_to_bf16(const float* __restrict__ W, bf16_t* __restrict__ Wt,
                                  int K, int N) {
  const size_t i = (size_t)blockIdx.x * blockDim.x + threadIdx.x;
  const size_t total = (size_t)K * N;
  if (i >= total) return;
  const int n = (int)(i / K);
  const int k = (int)(i % K);
  Wt[i] = (bf16_t)W[(size_t)k * N + n];
}

__global__ void init_h_kernel(const float* __restrict__ emb, const int* __restrict__ etype,
                              const int* __restrict__ mask, float* __restrict__ h,
                              bf16_t* __restrict__ h_bf) {
  const int row = blockIdx.x;      // 0..799
  const int tid = threadIdx.x;
  const int t = etype[row];
  const float m = mask[row] ? 1.0f : 0.0f;
  for (int j4 = 0; j4 < 4; ++j4) {
    const int d = tid + j4 * 256;
    const float v = emb[(size_t)t * D_ + d] * m;
    h[(size_t)row * D_ + d] = v;
    h_bf[(size_t)row * D_ + d] = (bf16_t)v;
  }
}

// ---------------------------------------------------------------------------
// Per-step kernels
// ---------------------------------------------------------------------------
__global__ void roles_kernel(const float* __restrict__ inc, const int* __restrict__ mask,
                             float* __restrict__ roles, float* __restrict__ wsum, int s) {
  __shared__ float red[128];
  const int b = blockIdx.x, tid = threadIdx.x;
  float r = 0.0f;
  if (tid < N_) {
    const int row = b * N_ + tid;
    r = inc[(size_t)row * S_ + s] * (mask[row] ? 1.0f : 0.0f);
    roles[row] = r;
  }
  red[tid] = r;
  __syncthreads();
  for (int st = 64; st > 0; st >>= 1) {
    if (tid < st) red[tid] += red[tid + st];
    __syncthreads();
  }
  if (tid == 0) wsum[b] = fmaxf(red[0], 1.0f);
}

__global__ void reduce_meanent(const float* __restrict__ roles, const float* __restrict__ wsum,
                               const float* __restrict__ h, float* __restrict__ tot,
                               float* __restrict__ mean_ent) {
  const int i = blockIdx.x * blockDim.x + threadIdx.x;   // 8192
  const int b = i >> 10, d = i & 1023;
  const float* hb = h + (size_t)b * N_ * D_ + d;
  const float* rb = roles + b * N_;
  float s1 = 0.0f;
  for (int n = 0; n < N_; ++n) s1 += rb[n] * hb[(size_t)n * D_];
  tot[i] = s1;
  mean_ent[i] = s1 / wsum[b];
}

// Split-K (8-way) partials for es_tmp = mean_ent@W_ne + b_ne + scene@W_te + b_te
__global__ void es_gemm_split(const float* __restrict__ mean_ent, const float* __restrict__ scene,
                              const float* __restrict__ Wne, const float* __restrict__ bne,
                              const float* __restrict__ Wte, const float* __restrict__ bte,
                              float* __restrict__ part, int s) {
  const int i = blockIdx.x * blockDim.x + threadIdx.x;   // 65536
  const int ks = i >> 13;             // 0..7 K-split slice
  const int j  = i & 8191;            // b*1024 + d
  const int b = j >> 10, d = j & 1023;
  const float* me = mean_ent + (size_t)b * D_;
  const float* sc = scene + ((size_t)b * S_ + s) * D_;
  float acc = (ks == 0) ? (bne[d] + bte[d]) : 0.0f;
  const int k0 = ks * 128;
  for (int k = k0; k < k0 + 128; ++k)
    acc += me[k] * Wne[(size_t)k * D_ + d] + sc[k] * Wte[(size_t)k * D_ + d];
  part[i] = acc;
}

// Split-K (8-way) partials for out = x @ W + b   (x is [8,1024])
__global__ void lin8_split(const float* __restrict__ x, const float* __restrict__ W,
                           const float* __restrict__ b, float* __restrict__ part) {
  const int i = blockIdx.x * blockDim.x + threadIdx.x;   // 65536
  const int ks = i >> 13;
  const int j  = i & 8191;
  const int bb = j >> 10, d = j & 1023;
  const float* xr = x + (size_t)bb * D_;
  float acc = (ks == 0) ? b[d] : 0.0f;
  const int k0 = ks * 128;
  for (int k = k0; k < k0 + 128; ++k) acc += xr[k] * W[(size_t)k * D_ + d];
  part[i] = acc;
}

// Deterministic fixed-order reduction of the 8 K-split partials.
__global__ void reduce8_kernel(const float* __restrict__ part, float* __restrict__ out) {
  const int j = blockIdx.x * blockDim.x + threadIdx.x;   // 8192
  float a = 0.0f;
#pragma unroll
  for (int ks = 0; ks < 8; ++ks) a += part[ks * 8192 + j];
  out[j] = a;
}

__device__ __forceinline__ float block_reduce256(float v, float* red) {
  const int tid = threadIdx.x;
  red[tid] = v;
  __syncthreads();
  for (int st = 128; st > 0; st >>= 1) {
    if (tid < st) red[tid] += red[tid + st];
    __syncthreads();
  }
  const float r = red[0];
  __syncthreads();
  return r;
}

__global__ void ln_es_kernel(const float* __restrict__ x, const float* __restrict__ g,
                             const float* __restrict__ bta, float* __restrict__ e_s,
                             float* __restrict__ buf, int s) {
  __shared__ float red[256];
  const int b = blockIdx.x, tid = threadIdx.x;
  const float* xr = x + (size_t)b * D_;
  float v[4], sum = 0.0f;
  for (int j = 0; j < 4; ++j) { v[j] = xr[tid + j * 256]; sum += v[j]; }
  const float mu = block_reduce256(sum, red) * (1.0f / D_);
  float s2 = 0.0f;
  for (int j = 0; j < 4; ++j) { const float dl = v[j] - mu; s2 += dl * dl; }
  const float var = block_reduce256(s2, red) * (1.0f / D_);
  const float inv = rsqrtf(var + 1e-5f);
  for (int j = 0; j < 4; ++j) {
    const int d = tid + j * 256;
    const float y = (v[j] - mu) * inv * g[d] + bta[d];
    e_s[(size_t)b * D_ + d] = y;
    buf[((size_t)b * S_ + s) * D_ + d] = y;
  }
}

// arc_feat (small-K contraction over past hyperedges) and social, written bf16
__global__ void prep_kernel(const float* __restrict__ inc, const float* __restrict__ roles,
                            const float* __restrict__ buf, const float* __restrict__ tot,
                            const float* __restrict__ h, bf16_t* __restrict__ arcA,
                            bf16_t* __restrict__ socA, int s) {
  __shared__ float cj[S_];
  const int row = blockIdx.x;     // 0..799
  const int b = row / N_;
  const int tid = threadIdx.x;
  const float role = roles[row];
  if (tid < S_) {
    float c = 0.0f;
    if (tid < s) {
      int dnm = s - 1; if (dnm < 1) dnm = 1;
      c = inc[(size_t)row * S_ + tid] * role * expf(-1.0f + (float)tid / (float)dnm);
    }
    cj[tid] = c;
  }
  __syncthreads();
  const float* bb = buf + (size_t)b * S_ * D_;
  for (int j4 = 0; j4 < 4; ++j4) {
    const int d = tid + j4 * 256;
    float af = 0.0f;
    for (int j = 0; j < s; ++j) af += cj[j] * bb[(size_t)j * D_ + d];
    arcA[(size_t)row * D_ + d] = (bf16_t)af;
    const float hv = h[(size_t)row * D_ + d];
    const float so = role * (tot[(size_t)b * D_ + d] - role * hv);
    socA[(size_t)row * D_ + d] = (bf16_t)so;
  }
}

// msg = w0*msg_scene + gate(s>0)*w1*(Y1+b_ma) + w2*(Y2+b_mi) -> bf16
__global__ void combine_kernel(const float* __restrict__ sm, const float* __restrict__ msg_sc,
                               const float* __restrict__ Y1, const float* __restrict__ bma,
                               const float* __restrict__ Y2, const float* __restrict__ bmi,
                               bf16_t* __restrict__ msgA, int s) {
  const int row = blockIdx.x, tid = threadIdx.x;
  const int b = row / N_;
  const float w0 = sm[0];
  const float w1 = (s > 0) ? sm[1] : 0.0f;
  const float w2 = sm[2];
  for (int j4 = 0; j4 < 4; ++j4) {
    const int d = tid + j4 * 256;
    const size_t idx = (size_t)row * D_ + d;
    const float m = w0 * msg_sc[(size_t)b * D_ + d] + w1 * (Y1[idx] + bma[d]) + w2 * (Y2[idx] + bmi[d]);
    msgA[idx] = (bf16_t)m;
  }
}

__global__ void gru_kernel(const float* __restrict__ gi, const float* __restrict__ gh,
                           const float* __restrict__ roles, float* __restrict__ h,
                           bf16_t* __restrict__ h_bf) {
  const int row = blockIdx.x, tid = threadIdx.x;
  const bool ins = roles[row] > 0.0f;
  const size_t base = (size_t)row * 3 * D_;
  for (int j4 = 0; j4 < 4; ++j4) {
    const int d = tid + j4 * 256;
    const float r = 1.0f / (1.0f + expf(-(gi[base + d] + gh[base + d])));
    const float z = 1.0f / (1.0f + expf(-(gi[base + D_ + d] + gh[base + D_ + d])));
    const float n_ = tanhf(gi[base + 2 * D_ + d] + r * gh[base + 2 * D_ + d]);
    const size_t hi = (size_t)row * D_ + d;
    const float hv = h[hi];
    const float hn = (1.0f - z) * n_ + z * hv;
    const float out = ins ? hn : hv;
    h[hi] = out;
    h_bf[hi] = (bf16_t)out;
  }
}

// ---------------------------------------------------------------------------
// Final-stage kernels
// ---------------------------------------------------------------------------
__global__ void resid_ln_kernel(const float* __restrict__ buf, const float* __restrict__ scene,
                                const float* __restrict__ g, const float* __restrict__ bta,
                                bf16_t* __restrict__ H1) {
  __shared__ float red[256];
  const int row = blockIdx.x;     // 0..383 (b*48+s)
  const int tid = threadIdx.x;
  float v[4], sum = 0.0f;
  for (int j = 0; j < 4; ++j) {
    const int d = tid + j * 256;
    v[j] = buf[(size_t)row * D_ + d] + scene[(size_t)row * D_ + d];
    sum += v[j];
  }
  const float mu = block_reduce256(sum, red) * (1.0f / D_);
  float s2 = 0.0f;
  for (int j = 0; j < 4; ++j) { const float dl = v[j] - mu; s2 += dl * dl; }
  const float var = block_reduce256(s2, red) * (1.0f / D_);
  const float inv = rsqrtf(var + 1e-5f);
  for (int j = 0; j < 4; ++j) {
    const int d = tid + j * 256;
    H1[(size_t)row * D_ + d] = (bf16_t)((v[j] - mu) * inv * g[d] + bta[d]);
  }
}

__device__ __forceinline__ float gelu_exact(float x) {
  return 0.5f * x * (1.0f + erff(x * 0.70710678118654752f));
}

__global__ void gelu_ln_out_kernel(const float* __restrict__ G, const float* __restrict__ g,
                                   const float* __restrict__ bta, float* __restrict__ out) {
  __shared__ float red[256];
  const int row = blockIdx.x;     // 0..383
  const int tid = threadIdx.x;
  float v[4], sum = 0.0f;
  for (int j = 0; j < 4; ++j) {
    v[j] = gelu_exact(G[(size_t)row * D_ + tid + j * 256]);
    sum += v[j];
  }
  const float mu = block_reduce256(sum, red) * (1.0f / D_);
  float s2 = 0.0f;
  for (int j = 0; j < 4; ++j) { const float dl = v[j] - mu; s2 += dl * dl; }
  const float var = block_reduce256(s2, red) * (1.0f / D_);
  const float inv = rsqrtf(var + 1e-5f);
  for (int j = 0; j < 4; ++j) {
    const int d = tid + j * 256;
    out[(size_t)row * D_ + d] = (v[j] - mu) * inv * g[d] + bta[d];
  }
}

__global__ void copy_h_kernel(const float* __restrict__ h, float* __restrict__ out) {
  const size_t i = (size_t)blockIdx.x * blockDim.x + threadIdx.x;  // 819200
  out[(size_t)BS_ * D_ + i] = h[i];
}

// ---------------------------------------------------------------------------
// Host orchestration
// ---------------------------------------------------------------------------
extern "C" void kernel_launch(void* const* d_in, const int* in_sizes, int n_in,
                              void* d_out, int out_size, void* d_ws, size_t ws_size,
                              hipStream_t stream) {
  (void)in_sizes; (void)n_in; (void)out_size; (void)ws_size;
  const float* emb     = (const float*)d_in[0];
  const float* W_ne    = (const float*)d_in[1];
  const float* b_ne    = (const float*)d_in[2];
  const float* W_te    = (const float*)d_in[3];
  const float* b_te    = (const float*)d_in[4];
  const float* ln_e_g  = (const float*)d_in[5];
  const float* ln_e_b  = (const float*)d_in[6];
  const float* stream_w= (const float*)d_in[7];
  const float* W_ms    = (const float*)d_in[8];
  const float* b_ms    = (const float*)d_in[9];
  const float* W_ma    = (const float*)d_in[10];
  const float* b_ma    = (const float*)d_in[11];
  const float* W_mi    = (const float*)d_in[12];
  const float* b_mi    = (const float*)d_in[13];
  const float* Wih     = (const float*)d_in[14];
  const float* Whh     = (const float*)d_in[15];
  const float* bih     = (const float*)d_in[16];
  const float* bhh     = (const float*)d_in[17];
  const float* ln_res_g= (const float*)d_in[18];
  const float* ln_res_b= (const float*)d_in[19];
  const float* W_eo    = (const float*)d_in[20];
  const float* b_eo    = (const float*)d_in[21];
  const float* ln_eo_g = (const float*)d_in[22];
  const float* ln_eo_b = (const float*)d_in[23];
  const float* scene   = (const float*)d_in[24];  // [8,48,1024]
  const float* inc     = (const float*)d_in[25];  // [8,100,48]
  // d_in[26] = edge_type_ids (unused by reference)
  const int*   etype   = (const int*)d_in[27];    // [8,100]
  const int*   emask   = (const int*)d_in[28];    // [8,100]

  char* ws = (char*)d_ws;
  size_t off = 0;
  auto alloc = [&](size_t bytes) {
    const size_t o = off;
    off = (off + bytes + 255) & ~(size_t)255;
    return o;
  };
  float*  stream_sm = (float*) (ws + alloc(4 * sizeof(float)));
  bf16_t* Wma_t = (bf16_t*)(ws + alloc((size_t)D_ * D_ * 2));
  bf16_t* Wmi_t = (bf16_t*)(ws + alloc((size_t)D_ * D_ * 2));
  bf16_t* Wih_t = (bf16_t*)(ws + alloc((size_t)3 * D_ * D_ * 2));
  bf16_t* Whh_t = (bf16_t*)(ws + alloc((size_t)3 * D_ * D_ * 2));
  bf16_t* Weo_t = (bf16_t*)(ws + alloc((size_t)D_ * D_ * 2));
  float*  hbuf  = (float*) (ws + alloc((size_t)BN_ * D_ * 4));
  bf16_t* h_bf  = (bf16_t*)(ws + alloc((size_t)BN_ * D_ * 2));
  float*  bufv  = (float*) (ws + alloc((size_t)B_ * S_ * D_ * 4));
  float*  rolesv= (float*) (ws + alloc((size_t)BN_ * 4));
  float*  wsumv = (float*) (ws + alloc((size_t)B_ * 4));
  float*  totv  = (float*) (ws + alloc((size_t)B_ * D_ * 4));
  float*  meanent=(float*) (ws + alloc((size_t)B_ * D_ * 4));
  float*  es_tmp= (float*) (ws + alloc((size_t)B_ * D_ * 4));
  float*  e_sv  = (float*) (ws + alloc((size_t)B_ * D_ * 4));
  float*  msg_sc= (float*) (ws + alloc((size_t)B_ * D_ * 4));
  float*  partv = (float*) (ws + alloc((size_t)8 * B_ * D_ * 4));
  bf16_t* arcA  = (bf16_t*)(ws + alloc((size_t)BN_ * D_ * 2));
  bf16_t* socA  = (bf16_t*)(ws + alloc((size_t)BN_ * D_ * 2));
  bf16_t* msgA  = (bf16_t*)(ws + alloc((size_t)BN_ * D_ * 2));
  float*  Y1    = (float*) (ws + alloc((size_t)BN_ * D_ * 4));
  float*  Y2    = (float*) (ws + alloc((size_t)BN_ * D_ * 4));
  float*  giv   = (float*) (ws + alloc((size_t)BN_ * 3 * D_ * 4));
  float*  ghv   = (float*) (ws + alloc((size_t)BN_ * 3 * D_ * 4));
  bf16_t* H1bf  = (bf16_t*)(ws + alloc((size_t)BS_ * D_ * 2));
  float*  Gv    = (float*) (ws + alloc((size_t)BS_ * D_ * 4));

  // Dual-problem GEMM launch: blockIdx.z in [0,nz) selects {A,B,bias,C}.
  auto gemm2 = [&](const bf16_t* A0, const bf16_t* A1,
                   const bf16_t* Bt0, const bf16_t* Bt1,
                   const float* bias0, const float* bias1,
                   float* C0, float* C1, int M, int N, int K, int nz) {
    dim3 grid(N / 64, (M + 127) / 128, nz);
    wmma_gemm_kernel<<<grid, dim3(128), 0, stream>>>(
        A0, A1, Bt0, Bt1, bias0, bias1, C0, C1, M, N, K);
  };

  // ---- setup (runs every call: deterministic, no caching) ----
  stream_softmax_kernel<<<1, 1, 0, stream>>>(stream_w, stream_sm);
  {
    const int t1 = (D_ * D_ + 255) / 256;
    const int t3 = (3 * D_ * D_ + 255) / 256;
    transpose_to_bf16<<<t1, 256, 0, stream>>>(W_ma, Wma_t, D_, D_);
    transpose_to_bf16<<<t1, 256, 0, stream>>>(W_mi, Wmi_t, D_, D_);
    transpose_to_bf16<<<t3, 256, 0, stream>>>(Wih,  Wih_t, D_, 3 * D_);
    transpose_to_bf16<<<t3, 256, 0, stream>>>(Whh,  Whh_t, D_, 3 * D_);
    transpose_to_bf16<<<t1, 256, 0, stream>>>(W_eo, Weo_t, D_, D_);
  }
  init_h_kernel<<<BN_, 256, 0, stream>>>(emb, etype, emask, hbuf, h_bf);

  // ---- sequential scan over hyperedges ----
  for (int s = 0; s < S_; ++s) {
    roles_kernel<<<B_, 128, 0, stream>>>(inc, emask, rolesv, wsumv, s);
    reduce_meanent<<<32, 256, 0, stream>>>(rolesv, wsumv, hbuf, totv, meanent);
    es_gemm_split<<<256, 256, 0, stream>>>(meanent, scene, W_ne, b_ne, W_te, b_te, partv, s);
    reduce8_kernel<<<32, 256, 0, stream>>>(partv, es_tmp);
    ln_es_kernel<<<B_, 256, 0, stream>>>(es_tmp, ln_e_g, ln_e_b, e_sv, bufv, s);
    lin8_split<<<256, 256, 0, stream>>>(e_sv, W_ms, b_ms, partv);
    reduce8_kernel<<<32, 256, 0, stream>>>(partv, msg_sc);
    prep_kernel<<<BN_, 256, 0, stream>>>(inc, rolesv, bufv, totv, hbuf, arcA, socA, s);
    gemm2(arcA, socA, Wma_t, Wmi_t, nullptr, nullptr, Y1, Y2, BN_, D_, D_, 2);
    combine_kernel<<<BN_, 256, 0, stream>>>(stream_sm, msg_sc, Y1, b_ma, Y2, b_mi, msgA, s);
    gemm2(msgA, h_bf, Wih_t, Whh_t, bih, bhh, giv, ghv, BN_, 3 * D_, D_, 2);
    gru_kernel<<<BN_, 256, 0, stream>>>(giv, ghv, rolesv, hbuf, h_bf);
  }

  // ---- final: H = LN(gelu(LN(buf+scene)@W_eo + b_eo)) ----
  float* outp = (float*)d_out;
  resid_ln_kernel<<<BS_, 256, 0, stream>>>(bufv, scene, ln_res_g, ln_res_b, H1bf);
  gemm2(H1bf, H1bf, Weo_t, Weo_t, b_eo, b_eo, Gv, Gv, BS_, D_, D_, 1);
  gelu_ln_out_kernel<<<BS_, 256, 0, stream>>>(Gv, ln_eo_g, ln_eo_b, outp);
  copy_h_kernel<<<(BN_ * D_) / 256, 256, 0, stream>>>(hbuf, outp);
}
